// Fuser_15702400434434
// MI455X (gfx1250) — matrix operand order repaired and checked by reading, not compiled, not run on hardware
//
#include <hip/hip_runtime.h>
#include <hip/hip_bf16.h>
#include <math.h>

// Problem sizes (fixed by the reference)
#define HIDDEN 1024
#define SENT_D 1024
#define NS 8192
#define NK 8192

typedef __bf16 bf16;
typedef __bf16 v16bf __attribute__((ext_vector_type(16)));
typedef __bf16 v8bf  __attribute__((ext_vector_type(8)));
typedef float  v8f   __attribute__((ext_vector_type(8)));
typedef float  v4f   __attribute__((ext_vector_type(4)));
typedef int    v4i   __attribute__((vector_size(16)));   // matches builtin param

// Block tile 128(M) x 256(N) x 32(K); 256 threads = 8 wave32 waves arranged
// 2(M) x 4(N); each wave computes a 64x64 tile = 4x4 v_wmma_f32_16x16x32_bf16.
#define BM 128
#define BN 256
#define BK 32

#if __has_builtin(__builtin_amdgcn_global_load_async_to_lds_b128) && \
    __has_builtin(__builtin_amdgcn_s_wait_asynccnt)
#define HAVE_ASYNC_LDS 1
#else
#define HAVE_ASYNC_LDS 0
#endif

// generic -> AS1 / AS3 pointer conversions (integer round-trip; AS3 address is
// the low 32 bits of a generic LDS pointer on amdgcn). Pointee type must be
// the b128 payload vector type expected by the builtin.
#define GPTR(p) ((__attribute__((address_space(1))) v4i*)(uintptr_t)(p))
#define LPTR(p) ((__attribute__((address_space(3))) v4i*)(uint32_t)(uintptr_t)(p))

__device__ __forceinline__ void async_cp16(const void* g, void* l) {
#if HAVE_ASYNC_LDS
  __builtin_amdgcn_global_load_async_to_lds_b128(GPTR(g), LPTR(l), 0, 0);
#else
  (void)g; (void)l;
#endif
}
template<int N>
__device__ __forceinline__ void wait_async() {
#if HAVE_ASYNC_LDS
  __builtin_amdgcn_s_wait_asynccnt(N);
#endif
}

__device__ __forceinline__ v16bf cat8(v8bf lo, v8bf hi) {
  return __builtin_shufflevector(lo, hi, 0,1,2,3,4,5,6,7,8,9,10,11,12,13,14,15);
}

// ---------------------------------------------------------------------------
// f32 -> bf16 packed conversion: 8 elements/thread (32B load, 16B store)
// ---------------------------------------------------------------------------
__global__ __launch_bounds__(256)
void cvt_f32_bf16(const float* __restrict__ src, bf16* __restrict__ dst, int n8) {
  int i = blockIdx.x * 256 + threadIdx.x;
  if (i >= n8) return;
  const v4f* s = (const v4f*)src + (size_t)i * 2;
  v4f a = s[0], b = s[1];
  v8bf o;
#pragma unroll
  for (int e = 0; e < 4; ++e) { o[e] = (bf16)a[e]; o[e + 4] = (bf16)b[e]; }
  ((v8bf*)dst)[i] = o;
}

// ---------------------------------------------------------------------------
// Tiled WMMA GEMM with double-buffered LDS pipeline.
//   B_NT = true : B source row-major [N,K]  (C = A * Bsrc^T)
//   B_NT = false: B source row-major [K,N]  (C = A * Bsrc), transposed into LDS
//   A_F32       : A is f32, converted to bf16 while staging
// Pure-bf16 NT case stages via GLOBAL_LOAD_ASYNC_TO_LDS_B128 (ASYNCcnt,
// ping-pong buffers); other cases use a register double buffer.
// ---------------------------------------------------------------------------
template<bool A_F32, bool HAS_BIAS, bool OUT_BF16, bool B_NT>
__global__ __launch_bounds__(256)
void gemm_wmma(const void* __restrict__ Aptr, const bf16* __restrict__ B,
               const float* __restrict__ bias, void* __restrict__ Cptr,
               int M, int N, int K)
{
  __shared__ bf16 As[2][BM * BK];   // 2 x 8 KB
  __shared__ bf16 Bs[2][BN * BK];   // 2 x 16 KB

  const int t    = threadIdx.x;
  const int lane = t & 31;
  const int wave = t >> 5;
  const int wm   = wave >> 2;      // 0..1 -> M offset wm*64
  const int wn   = wave & 3;       // 0..3 -> N offset wn*64
  const int m0   = blockIdx.y * BM;
  const int n0   = blockIdx.x * BN;

  const int mfrag = lane & 15;
  const int hi16  = lane >> 4;     // 0|1
  const int akb   = hi16 * 8;      // ISA 16-bit A 16x32 lane layout
  const int bkb   = hi16 * 8;      // B k order matches A (table order)

  v8f zero = {};
  v8f acc[4][4];
#pragma unroll
  for (int i = 0; i < 4; ++i)
#pragma unroll
    for (int j = 0; j < 4; ++j) acc[i][j] = zero;

  auto compute = [&](int buf) {
    v16bf afrag[4];
#pragma unroll
    for (int i = 0; i < 4; ++i) {
      int row = wm * 64 + i * 16 + mfrag;
      v8bf lo = *(const v8bf*)&As[buf][row * BK + akb];
      v8bf hi = *(const v8bf*)&As[buf][row * BK + akb + 16];
      afrag[i] = cat8(lo, hi);
    }
#pragma unroll
    for (int j = 0; j < 4; ++j) {
      int col = wn * 64 + j * 16 + mfrag;
      v8bf lo = *(const v8bf*)&Bs[buf][col * BK + bkb];
      v8bf hi = *(const v8bf*)&Bs[buf][col * BK + bkb + 16];
      v16bf bfragj = cat8(lo, hi);
#pragma unroll
      for (int i = 0; i < 4; ++i)
        acc[i][j] = __builtin_amdgcn_wmma_f32_16x16x32_bf16(
            false, afrag[i], false, bfragj, (short)0, acc[i][j], false, false);
    }
  };

  constexpr bool USE_ASYNC = (HAVE_ASYNC_LDS != 0) && !A_F32 && B_NT;

  if constexpr (USE_ASYNC) {
    const bf16* Ab = (const bf16*)Aptr;
    auto issue = [&](int buf, int k0) {
      // A tile: BM*BK = 4096 bf16 -> 2 x b128 per thread
#pragma unroll
      for (int rep = 0; rep < 2; ++rep) {
        int idx = t * 8 + rep * 2048;
        int r = idx >> 5, c = idx & 31;
        async_cp16(Ab + (size_t)(m0 + r) * K + k0 + c, &As[buf][idx]);
      }
      // B tile: BN*BK = 8192 bf16 -> 4 x b128 per thread
#pragma unroll
      for (int rep = 0; rep < 4; ++rep) {
        int idx = t * 8 + rep * 2048;
        int r = idx >> 5, c = idx & 31;
        async_cp16(B + (size_t)(n0 + r) * K + k0 + c, &Bs[buf][idx]);
      }
    };
    issue(0, 0);
    int buf = 0;
    for (int k0 = 0; k0 < K; k0 += BK) {
      if (k0 + BK < K) { issue(buf ^ 1, k0 + BK); wait_async<6>(); }
      else             { wait_async<0>(); }
      __syncthreads();          // all waves' async writes for `buf` landed
      compute(buf);
      __syncthreads();          // safe to overwrite `buf` next iteration
      buf ^= 1;
    }
  } else {
    // register double buffer: loads for stage k+1 issued before compute(k)
    v4f  aregF[2][2];           // A_F32 raw
    v8bf aregB[2];              // A bf16 raw
    v8bf breg[4];               // B raw
    auto loadA = [&](int k0) {
#pragma unroll
      for (int rep = 0; rep < 2; ++rep) {
        int idx = t * 8 + rep * 2048;
        int r = idx >> 5, c = idx & 31;
        if constexpr (A_F32) {
          const float* Af = (const float*)Aptr;
          const v4f* gp = (const v4f*)(Af + (size_t)(m0 + r) * K + k0 + c);
          aregF[rep][0] = gp[0];
          aregF[rep][1] = gp[1];
        } else {
          const bf16* Ab = (const bf16*)Aptr;
          aregB[rep] = *(const v8bf*)(Ab + (size_t)(m0 + r) * K + k0 + c);
        }
      }
    };
    auto loadB = [&](int k0) {
#pragma unroll
      for (int rep = 0; rep < 4; ++rep) {
        int idx = t * 8 + rep * 2048;
        if constexpr (B_NT) {
          int r = idx >> 5, c = idx & 31;
          breg[rep] = *(const v8bf*)(B + (size_t)(n0 + r) * K + k0 + c);
        } else {
          int nn = idx & 255, kk = idx >> 8;
          breg[rep] = *(const v8bf*)(B + (size_t)(k0 + kk) * N + n0 + nn);
        }
      }
    };
    auto store = [&](int buf) {
#pragma unroll
      for (int rep = 0; rep < 2; ++rep) {
        int idx = t * 8 + rep * 2048;
        v8bf av;
        if constexpr (A_F32) {
#pragma unroll
          for (int e = 0; e < 4; ++e) {
            av[e]     = (bf16)aregF[rep][0][e];
            av[e + 4] = (bf16)aregF[rep][1][e];
          }
        } else {
          av = aregB[rep];
        }
        *(v8bf*)&As[buf][idx] = av;
      }
#pragma unroll
      for (int rep = 0; rep < 4; ++rep) {
        int idx = t * 8 + rep * 2048;
        if constexpr (B_NT) {
          *(v8bf*)&Bs[buf][idx] = breg[rep];
        } else {
          int nn = idx & 255, kk = idx >> 8;
#pragma unroll
          for (int e = 0; e < 8; ++e) Bs[buf][(nn + e) * BK + kk] = breg[rep][e];
        }
      }
    };
    loadA(0); loadB(0);
    int buf = 0;
    for (int k0 = 0; k0 < K; k0 += BK) {
      store(buf);
      if (k0 + BK < K) { loadA(k0 + BK); loadB(k0 + BK); }
      __syncthreads();
      compute(buf);
      __syncthreads();
      buf ^= 1;
    }
  }

  // epilogue: C/D layout -> lane n = lane&15, VGPR r holds M = r + hi16*8
  bf16*  Cb = (bf16*)Cptr;
  float* Cf = (float*)Cptr;
#pragma unroll
  for (int i = 0; i < 4; ++i) {
    int rbase = m0 + wm * 64 + i * 16 + hi16 * 8;
#pragma unroll
    for (int j = 0; j < 4; ++j) {
      int col = n0 + wn * 64 + j * 16 + mfrag;
      float bv = HAS_BIAS ? bias[col] : 0.0f;
#pragma unroll
      for (int r = 0; r < 8; ++r) {
        float v = acc[i][j][r] + bv;
        size_t o = (size_t)(rbase + r) * N + col;
        if (OUT_BF16) Cb[o] = (bf16)v; else Cf[o] = v;
      }
    }
  }
}

// ---------------------------------------------------------------------------
// Column softmax (axis=0) in place; optionally also writes a bf16 copy.
// Block = 1024 threads = 64 columns x 16 row-groups, coalesced row reads.
// ---------------------------------------------------------------------------
__global__ __launch_bounds__(1024)
void softmax_col(float* __restrict__ sc, bf16* __restrict__ scb, int Ns, int Nk) {
  const int COLS = 64, GROUPS = 16;
  __shared__ float sm[GROUPS][COLS];
  __shared__ float sl[GROUPS][COLS];

  int t = threadIdx.x;
  int c = t & (COLS - 1);
  int g = t >> 6;
  int col = blockIdx.x * COLS + c;
  int rows_per = Ns / GROUPS;
  int r0 = g * rows_per;

  float m = -INFINITY, l = 0.0f;
  for (int r = r0; r < r0 + rows_per; ++r) {
    float x = sc[(size_t)r * Nk + col];
    float nm = fmaxf(m, x);
    l = l * __expf(m - nm) + __expf(x - nm);
    m = nm;
  }
  sm[g][c] = m; sl[g][c] = l;
  __syncthreads();
  if (g == 0) {
    float M = sm[0][c], L = sl[0][c];
#pragma unroll
    for (int gg = 1; gg < GROUPS; ++gg) {
      float m2 = sm[gg][c];
      float nm = fmaxf(M, m2);
      L = L * __expf(M - nm) + sl[gg][c] * __expf(m2 - nm);
      M = nm;
    }
    sm[0][c] = M; sl[0][c] = L;
  }
  __syncthreads();
  float M = sm[0][c];
  float inv = 1.0f / sl[0][c];
  for (int r = r0; r < r0 + rows_per; ++r) {
    size_t o = (size_t)r * Nk + col;
    float v = __expf(sc[o] - M) * inv;
    sc[o] = v;
    if (scb) scb[o] = (bf16)v;
  }
}

// ---------------------------------------------------------------------------
// Launch sequence
// ---------------------------------------------------------------------------
extern "C" void kernel_launch(void* const* d_in, const int* in_sizes, int n_in,
                              void* d_out, int out_size, void* d_ws, size_t ws_size,
                              hipStream_t stream) {
  const float* sent = (const float*)d_in[0];   // [NS, SENT_D]
  const float* know = (const float*)d_in[1];   // [NK, 2*HIDDEN]
  const float* Ws   = (const float*)d_in[2];   // [HIDDEN, SENT_D]
  const float* bs   = (const float*)d_in[3];
  const float* Wk   = (const float*)d_in[4];   // [HIDDEN, 2*HIDDEN]
  const float* bk   = (const float*)d_in[5];

  float* attns = (float*)d_out;                          // [NS, NK]
  float* fused = attns + (size_t)NS * NK;                // [NS, 2*HIDDEN]

  // Workspace layout (bf16 buffers)
  char* ws = (char*)d_ws;
  bf16* sent_b = (bf16*)(ws + 0);                        // 16 MB
  bf16* Ws_b   = (bf16*)(ws + 16777216ull);              //  2 MB
  bf16* know_b = (bf16*)(ws + 18874368ull);              // 32 MB (also = V)
  bf16* Wk_b   = (bf16*)(ws + 52428800ull);              //  4 MB
  bf16* S_b    = (bf16*)(ws + 56623104ull);              // 16 MB
  bf16* K_b    = (bf16*)(ws + 73400320ull);              // 16 MB
  const size_t base_need = 90177536ull;
  // optional bf16 attns (128 MB) halves the dominant HBM stream of the step-6 GEMM
  bf16* attns_b = (ws_size >= base_need + ((size_t)NS * NK * 2))
                      ? (bf16*)(ws + base_need) : nullptr;

  // 1) convert inputs to bf16
  {
    int n8;
    n8 = NS * SENT_D / 8;         cvt_f32_bf16<<<n8 / 256, 256, 0, stream>>>(sent, sent_b, n8);
    n8 = HIDDEN * SENT_D / 8;     cvt_f32_bf16<<<n8 / 256, 256, 0, stream>>>(Ws,   Ws_b,   n8);
    n8 = NK * 2 * HIDDEN / 8;     cvt_f32_bf16<<<n8 / 256, 256, 0, stream>>>(know, know_b, n8);
    n8 = HIDDEN * 2 * HIDDEN / 8; cvt_f32_bf16<<<n8 / 256, 256, 0, stream>>>(Wk,   Wk_b,   n8);
  }

  // 2) S = sent @ Ws^T + bs   [NS, HIDDEN] bf16
  gemm_wmma<false, true, true, true>
      <<<dim3(HIDDEN / BN, NS / BM), 256, 0, stream>>>(sent_b, Ws_b, bs, S_b,
                                                       NS, HIDDEN, SENT_D);
  // 3) K = know @ Wk^T + bk   [NK, HIDDEN] bf16
  gemm_wmma<false, true, true, true>
      <<<dim3(HIDDEN / BN, NK / BM), 256, 0, stream>>>(know_b, Wk_b, bk, K_b,
                                                       NK, HIDDEN, 2 * HIDDEN);
  // 4) scores = S @ K^T       [NS, NK] f32 -> d_out (S,K bf16 are L2-resident)
  gemm_wmma<false, false, false, true>
      <<<dim3(NK / BN, NS / BM), 256, 0, stream>>>(S_b, K_b, nullptr, attns,
                                                   NS, NK, HIDDEN);
  // 5) softmax over axis=0, in place (+ optional bf16 copy)
  softmax_col<<<NK / 64, 1024, 0, stream>>>(attns, attns_b, NS, NK);

  // 6) fused = attns @ V      [NS, 2*HIDDEN] f32
  if (attns_b) {
    gemm_wmma<false, false, false, false>
        <<<dim3(2 * HIDDEN / BN, NS / BM), 256, 0, stream>>>(attns_b, know_b, nullptr,
                                                             fused, NS, 2 * HIDDEN, NK);
  } else {
    gemm_wmma<true, false, false, false>
        <<<dim3(2 * HIDDEN / BN, NS / BM), 256, 0, stream>>>(attns, know_b, nullptr,
                                                             fused, NS, 2 * HIDDEN, NK);
  }
}
